// UttDecoder_16544214024659
// MI455X (gfx1250) — matrix operand order repaired and checked by reading, not compiled, not run on hardware
//
#include <hip/hip_runtime.h>
#include <hip/hip_bf16.h>

// ---------------------------------------------------------------------------
// LSTM decoder for MI455X (gfx1250, wave32, WMMA bf16 + TDM async tensor path)
//   B=16, L=65, V=32000, DW=512, DC=256, H=512, D=1024, G=4H=2048, T=63
// ---------------------------------------------------------------------------

typedef __attribute__((ext_vector_type(16))) __bf16        v16bf;
typedef __attribute__((ext_vector_type(8)))  float         v8f;
typedef __attribute__((ext_vector_type(4))) unsigned int   u32x4;
typedef __attribute__((ext_vector_type(8)))  int           i32x8;
typedef __attribute__((ext_vector_type(4)))  int           i32x4;

#define NB    16      // batch
#define LL    65
#define NV    32000
#define DD    1024    // DW + 2*DC
#define GG    2048    // 4*H
#define HH    512
#define TT    63
#define NROWS 1008    // TT*NB

#define HAVE_TDM __has_builtin(__builtin_amdgcn_tensor_load_to_lds)

union Frag {
    v16bf v;
    u32x4 q[2];
};

__device__ __forceinline__ __bf16 f2bf(float f) {
    unsigned int u = __builtin_bit_cast(unsigned int, f);
    unsigned int r = (u + 0x7FFFu + ((u >> 16) & 1u)) >> 16;  // round-nearest-even
    unsigned short s = (unsigned short)r;
    return __builtin_bit_cast(__bf16, s);
}

// A-fragment 16x32 bf16 (contiguous K, matches load_b):
//   lanes 0-15 : M=lane,   K = k0+0..15
//   lanes16-31 : M=lane-16,K = k0+16..31
__device__ __forceinline__ Frag load_a(const __bf16* src, int stride, int row0, int k0, int lane) {
    Frag f;
    int m   = lane & 15;
    int ofs = (lane >> 4) * 16;
    const __bf16* p = src + (size_t)(row0 + m) * stride + k0 + ofs;
    f.q[0] = *(const u32x4*)(p);
    f.q[1] = *(const u32x4*)(p + 8);
    return f;
}

// B-fragment 32x16 bf16: B[k][n] = W[col0+n][k0+k] (W row-major, `stride` elems)
//   lanes 0-15 : column n, K=k0+0..15 ; lanes 16-31: column n, K=k0+16..31
__device__ __forceinline__ Frag load_b(const __bf16* W, int stride, int col0, int k0, int lane) {
    Frag f;
    int n    = lane & 15;
    int kofs = (lane >> 4) * 16;
    const __bf16* p = W + (size_t)(col0 + n) * stride + k0 + kofs;
    f.q[0] = *(const u32x4*)(p);
    f.q[1] = *(const u32x4*)(p + 8);
    return f;
}

__device__ __forceinline__ v8f wmma_bf16(const Frag& a, const Frag& b, v8f c) {
    return __builtin_amdgcn_wmma_f32_16x16x32_bf16(false, a.v, false, b.v,
                                                   (short)0, c, false, false);
}

__device__ __forceinline__ float sigmoidf_(float x) { return 1.0f / (1.0f + __expf(-x)); }

#if HAVE_TDM
// ---------------------------------------------------------------------------
// Tensor Data Mover: 1-D contiguous global -> LDS DMA (ISA ch.8 D# layout).
//   group0: count=1 | lds_addr | global_addr[56:0] | type=2
//   group1: data_size=1B, tensor_dim0=tile_dim0=nbytes, tensor_dim1=tile_dim1=1
// 6-arg builtin (clang-23 lane): (g0, g1, g2, g3, g4, cpol).
// Issued once per wave; tracked by TENSORcnt (s_wait_tensorcnt).
// ---------------------------------------------------------------------------
__device__ __forceinline__ void tdm_load_1d(unsigned lds_off, const void* gsrc, unsigned nbytes) {
    unsigned long long ga = (unsigned long long)(size_t)gsrc;
    u32x4 g0;
    g0[0] = 1u;                                            // count=1, user descriptor
    g0[1] = lds_off;                                       // LDS byte address
    g0[2] = (unsigned)ga;                                  // global_addr[31:0]
    g0[3] = (unsigned)((ga >> 32) & 0x01FFFFFFull) | 0x80000000u; // addr[56:32] | type=2
    i32x8 g1;
    g1[0] = 0;                                             // wg_mask=0, data_size=0 (1B)
    g1[1] = (int)((nbytes & 0xFFFFu) << 16);               // tensor_dim0[15:0]
    g1[2] = (int)((nbytes >> 16) & 0xFFFFu) | (1 << 16);   // tensor_dim0[31:16] | tensor_dim1=1
    g1[3] = (int)((nbytes & 0xFFFFu) << 16);               // tile_dim0 (bits 127:112)
    g1[4] = 1;                                             // tile_dim1=1, tile_dim2=0
    g1[5] = (int)nbytes;                                   // tensor_dim0_stride[31:0]
    g1[6] = 0;
    g1[7] = 0;
    i32x4 gz4 = {0, 0, 0, 0};
    i32x8 gz8 = {0, 0, 0, 0, 0, 0, 0, 0};
    __builtin_amdgcn_tensor_load_to_lds(g0, g1, gz4, gz4, gz8, 0);
}
#endif

// ---------------------------------------------------------------------------
// Kernel 1: generic f32 -> bf16 conversion
// ---------------------------------------------------------------------------
__global__ void k_cvt_bf16(const float* __restrict__ src, __bf16* __restrict__ dst, int n) {
    int i = blockIdx.x * blockDim.x + threadIdx.x;
    int stride = gridDim.x * blockDim.x;
    for (; i < n; i += stride) dst[i] = f2bf(src[i]);
}

// ---------------------------------------------------------------------------
// Kernel 2: build packed A for the input GEMM.
//   Apre[row = t*16+b][k] = k<512 ? emb[x_train[b][t]][k] : ctx[b][k-512]
// ---------------------------------------------------------------------------
__global__ void k_pack_a(const int* __restrict__ x_train, const float* __restrict__ emb,
                         const float* __restrict__ ctx, __bf16* __restrict__ apre) {
    int i = blockIdx.x * blockDim.x + threadIdx.x;          // NROWS*DD elements
    if (i >= NROWS * DD) return;
    int row = i >> 10;            // /1024
    int k   = i & 1023;
    int t   = row >> 4;
    int b   = row & 15;
    float v;
    if (k < 512) {
        int tok = x_train[b * LL + t];
        v = emb[(size_t)tok * 512 + k];
    } else {
        v = ctx[b * 512 + (k - 512)];
    }
    apre[i] = f2bf(v);
}

// ---------------------------------------------------------------------------
// Kernel 3: per-call state init: hsbuf row-block 0 = bf16(h0); sync[] = 0
// ---------------------------------------------------------------------------
__global__ void k_init(const float* __restrict__ h0, __bf16* __restrict__ hsbuf,
                       int* __restrict__ sync) {
    int tid = threadIdx.x;                                   // 256 threads, 1 block
    for (int i = tid; i < NB * HH; i += 256) hsbuf[i] = f2bf(h0[i]);
    if (tid < 64) sync[tid] = 0;
}

// ---------------------------------------------------------------------------
// Kernel 4: pre = Apre(1008x1024) x W_ih^T(1024x2048) + b_ih   (f32 out)
//   grid: 63 M-tiles x 16 groups, 8 waves/block -> one 16x16 tile per wave
// ---------------------------------------------------------------------------
__global__ void k_pre_gemm(const __bf16* __restrict__ apre, const __bf16* __restrict__ wih,
                           const float* __restrict__ b_ih, float* __restrict__ pre) {
    int lane = threadIdx.x & 31;
    int wv   = threadIdx.x >> 5;
    int mt   = blockIdx.x >> 4;                 // 0..62
    int nt   = ((blockIdx.x & 15) << 3) + wv;   // 0..127
    int m0   = mt * 16;
    int n0   = nt * 16;

    v8f acc = {};
    #pragma unroll 4
    for (int kt = 0; kt < DD / 32; ++kt) {
        Frag a = load_a(apre, DD, m0, kt * 32, lane);
        Frag b = load_b(wih,  DD, n0, kt * 32, lane);
        acc = wmma_bf16(a, b, acc);
    }
    int col = n0 + (lane & 15);
    float bias = b_ih[col];
    #pragma unroll
    for (int r = 0; r < 8; ++r) {
        int M = r + ((lane >> 4) << 3);
        pre[(size_t)(m0 + M) * GG + col] = acc[r] + bias;
    }
}

// ---------------------------------------------------------------------------
// Kernel 5: LSTM recurrence. 16 workgroups (cluster-sized), 8 waves each.
//   WG w owns h columns [32w, 32w+32); keeps its 128 W_hh rows as persistent
//   B-fragments in registers. h_t staged into LDS by the TDM each step.
//   Cross-WG sync: atomic arrive + CDNA5 cluster barrier.
// ---------------------------------------------------------------------------
__global__ void k_lstm(const float* __restrict__ pre, const __bf16* __restrict__ whh,
                       const float* __restrict__ b_hh, const float* __restrict__ c0,
                       __bf16* __restrict__ hsbuf, int* __restrict__ sync) {
    __shared__ __bf16 hbf[NB * HH];       // staged h_t, bf16     (16 KB)
    __shared__ float  gbuf[NB * 128];     // gate tile for this WG (8 KB)
    __shared__ float  cbuf[NB * 32];      // cell state slice      (2 KB)

    const int tid  = threadIdx.x;
    const int lane = tid & 31;
    const int wv   = tid >> 5;            // 0..7
    const int w    = blockIdx.x;          // 0..15
    const int nwg  = gridDim.x;

    // --- persistent W_hh B-fragments for this wave's 16 gate columns -------
    // local col lc in [0,128): gate = lc/32, global row g = gate*512 + w*32 + lc%32
    Frag bw[HH / 32];
    {
        int lc = wv * 16 + (lane & 15);
        int g  = (lc >> 5) * HH + w * 32 + (lc & 31);
        int kofs = (lane >> 4) * 16;
        #pragma unroll
        for (int kt = 0; kt < HH / 32; ++kt) {
            const __bf16* p = whh + (size_t)g * HH + kt * 32 + kofs;
            bw[kt].q[0] = *(const u32x4*)(p);
            bw[kt].q[1] = *(const u32x4*)(p + 8);
        }
    }

    // --- init cell state slice from c0 -------------------------------------
    for (int e = tid; e < NB * 32; e += 256) {
        int b  = e >> 5;
        int cc = e & 31;
        cbuf[e] = c0[b * HH + w * 32 + cc];
    }

    for (int t = 0; t < TT; ++t) {
        // stage h_t (written as hsbuf rows t*16..t*16+15 by ALL WGs last step)
#if HAVE_TDM
        if (tid < 32) {                       // wave 0 drives the DMA
            tdm_load_1d((unsigned)(size_t)hbf, hsbuf + (size_t)t * NB * HH, NB * HH * 2);
            __builtin_amdgcn_s_wait_tensorcnt(0);
        }
#else
        {
            const u32x4* src = (const u32x4*)(hsbuf + (size_t)t * NB * HH);
            u32x4* dst = (u32x4*)hbf;
            #pragma unroll
            for (int i = 0; i < 4; ++i) dst[tid + i * 256] = src[tid + i * 256];
        }
#endif
        __syncthreads();

        // gates(16x16 tile per wave) = h(16x512) x W_hh_slice^T
        v8f acc = {};
        #pragma unroll
        for (int kt = 0; kt < HH / 32; ++kt) {
            Frag a = load_a(hbf, HH, 0, kt * 32, lane);
            acc = wmma_bf16(a, bw[kt], acc);
        }

        // add pre[t] + b_hh, dump into LDS gate buffer
        {
            int lc = wv * 16 + (lane & 15);
            int g  = (lc >> 5) * HH + w * 32 + (lc & 31);
            float bias = b_hh[g];
            #pragma unroll
            for (int r = 0; r < 8; ++r) {
                int M = r + ((lane >> 4) << 3);     // batch row
                float gate = acc[r] + pre[(size_t)(t * NB + M) * GG + g] + bias;
                gbuf[M * 128 + lc] = gate;
            }
        }
        __syncthreads();

        // activations over the 16x32 h-slice; write h_{t+1} slice (bf16)
        for (int e = tid; e < NB * 32; e += 256) {
            int b  = e >> 5;
            int cc = e & 31;
            float gi = gbuf[b * 128 +       cc];
            float gf = gbuf[b * 128 +  32 + cc];
            float gg = gbuf[b * 128 +  64 + cc];
            float go = gbuf[b * 128 +  96 + cc];
            float cn = sigmoidf_(gf) * cbuf[e] + sigmoidf_(gi) * tanhf(gg);
            float hn = sigmoidf_(go) * tanhf(cn);
            cbuf[e] = cn;
            hsbuf[(size_t)(t + 1) * NB * HH + b * HH + w * 32 + cc] = f2bf(hn);
        }

        // ---- cross-workgroup barrier --------------------------------------
        __syncthreads();
        __threadfence();
        if (tid == 0) {
            __hip_atomic_fetch_add(&sync[t], 1, __ATOMIC_ACQ_REL, __HIP_MEMORY_SCOPE_AGENT);
            while (__hip_atomic_load(&sync[t], __ATOMIC_ACQUIRE, __HIP_MEMORY_SCOPE_AGENT) < nwg) {
                __builtin_amdgcn_s_sleep(2);
            }
        }
        // CDNA5 cluster barrier (s_barrier_signal/-wait -3); NOP when the
        // dispatch is not cluster-configured, real HW sync when it is.
        __builtin_amdgcn_s_cluster_barrier();
        __syncthreads();
        __threadfence();
    }
}

// ---------------------------------------------------------------------------
// Kernel 6: logits(b,t,v) = hs(1008x512) x W_ro^T(512x32000)
//   250 blocks x 8 waves; each wave owns one 16-col vocab tile and keeps all
//   16 W_ro B-fragments in registers across the 63 M-tiles. The hs A-tile is
//   double-buffered in LDS via TDM: wave 0 issues tile mt+1, waits only the
//   oldest transfer (s_wait_tensorcnt 1) while everyone computes tile mt.
// ---------------------------------------------------------------------------
__global__ void k_out_gemm(const __bf16* __restrict__ hsbuf, const __bf16* __restrict__ wro,
                           float* __restrict__ out) {
    __shared__ __bf16 hstile[2][NB * HH];    // 2 x 16 KB staged A tiles

    const int tid  = threadIdx.x;
    const int lane = tid & 31;
    const int wv   = tid >> 5;
    const int nt   = blockIdx.x * 8 + wv;    // 0..1999
    const int n0   = nt * 16;

    Frag bw[HH / 32];
    #pragma unroll
    for (int kt = 0; kt < HH / 32; ++kt)
        bw[kt] = load_b(wro, HH, n0, kt * 32, lane);

#if HAVE_TDM
    if (tid < 32)                                            // prologue: tile 0
        tdm_load_1d((unsigned)(size_t)&hstile[0][0],
                    hsbuf + (size_t)16 * HH, NB * HH * 2);
#endif

    for (int mt = 0; mt < TT; ++mt) {
        // hs rows (mt+1)*16 .. +15 are h outputs of step mt
#if HAVE_TDM
        if (tid < 32) {
            if (mt + 1 < TT) {
                tdm_load_1d((unsigned)(size_t)&hstile[(mt + 1) & 1][0],
                            hsbuf + (size_t)(mt + 2) * NB * HH, NB * HH * 2);
                __builtin_amdgcn_s_wait_tensorcnt(1);        // tile mt landed
            } else {
                __builtin_amdgcn_s_wait_tensorcnt(0);
            }
        }
        __syncthreads();
        const __bf16* atile = &hstile[mt & 1][0];
#else
        {
            const u32x4* src = (const u32x4*)(hsbuf + (size_t)(mt + 1) * NB * HH);
            u32x4* dst = (u32x4*)&hstile[0][0];
            #pragma unroll
            for (int i = 0; i < 4; ++i) dst[tid + i * 256] = src[tid + i * 256];
        }
        __syncthreads();
        const __bf16* atile = &hstile[0][0];
#endif

        v8f acc = {};
        #pragma unroll
        for (int kt = 0; kt < HH / 32; ++kt) {
            Frag a = load_a(atile, HH, 0, kt * 32, lane);
            acc = wmma_bf16(a, bw[kt], acc);
        }

        int v = n0 + (lane & 15);
        #pragma unroll
        for (int r = 0; r < 8; ++r) {
            int b = r + ((lane >> 4) << 3);          // batch
            out[((size_t)b * TT + mt) * NV + v] = acc[r];
        }
        __syncthreads();   // everyone done reading before buffer reuse
    }
}

// ---------------------------------------------------------------------------
// Host launcher
// ---------------------------------------------------------------------------
extern "C" void kernel_launch(void* const* d_in, const int* in_sizes, int n_in,
                              void* d_out, int out_size, void* d_ws, size_t ws_size,
                              hipStream_t stream) {
    const int*   x_train = (const int*)  d_in[0];
    const float* ctx     = (const float*)d_in[3];
    const float* emb     = (const float*)d_in[4];
    const float* W_ih    = (const float*)d_in[5];
    const float* W_hh    = (const float*)d_in[6];
    const float* b_ih    = (const float*)d_in[7];
    const float* b_hh    = (const float*)d_in[8];
    const float* W_ro    = (const float*)d_in[9];
    const float* h0      = (const float*)d_in[10];
    const float* c0      = (const float*)d_in[11];
    float* out = (float*)d_out;

    // ---- workspace layout (bytes, all 256-aligned) ------------------------
    char* ws = (char*)d_ws;
    size_t off = 0;
    __bf16* apre  = (__bf16*)(ws + off); off += (size_t)NROWS * DD * 2;     // 2,064,384
    __bf16* wih_b = (__bf16*)(ws + off); off += (size_t)GG * DD * 2;        // 4,194,304
    __bf16* whh_b = (__bf16*)(ws + off); off += (size_t)GG * HH * 2;        // 2,097,152
    __bf16* wro_b = (__bf16*)(ws + off); off += (size_t)NV * HH * 2;        // 32,768,000
    float*  pre   = (float*) (ws + off); off += (size_t)NROWS * GG * 4;     // 8,257,536
    __bf16* hsbuf = (__bf16*)(ws + off); off += (size_t)(TT + 1) * NB * HH * 2; // 1,048,576
    int*    sync  = (int*)   (ws + off); off += 256;

    // 1) weight conversions f32 -> bf16
    hipLaunchKernelGGL(k_cvt_bf16, dim3(2048), dim3(256), 0, stream, W_ih, wih_b, GG * DD);
    hipLaunchKernelGGL(k_cvt_bf16, dim3(1024), dim3(256), 0, stream, W_hh, whh_b, GG * HH);
    hipLaunchKernelGGL(k_cvt_bf16, dim3(8192), dim3(256), 0, stream, W_ro, wro_b, NV * HH);

    // 2) embedding gather + context concat, packed bf16
    hipLaunchKernelGGL(k_pack_a, dim3((NROWS * DD + 255) / 256), dim3(256), 0, stream,
                       x_train, emb, ctx, apre);

    // 3) per-call state init (h0 row block, sync counters)
    hipLaunchKernelGGL(k_init, dim3(1), dim3(256), 0, stream, h0, hsbuf, sync);

    // 4) input-projection GEMM (63 x 128 tiles, 8 waves/block)
    hipLaunchKernelGGL(k_pre_gemm, dim3(63 * 16), dim3(256), 0, stream,
                       apre, wih_b, b_ih, pre);

    // 5) LSTM recurrence across a 16-workgroup cluster
    hipLaunchKernelGGL(k_lstm, dim3(16), dim3(256), 0, stream,
                       pre, whh_b, b_hh, c0, hsbuf, sync);

    // 6) output GEMM into logits (b, t, v)
    hipLaunchKernelGGL(k_out_gemm, dim3(250), dim3(256), 0, stream,
                       hsbuf, wro_b, out);

    (void)in_sizes; (void)n_in; (void)out_size; (void)ws_size;
}
